// qcGEM_Decoder_18854906429829
// MI455X (gfx1250) — compile-verified, matrix operand order checked by reading, not run on hardware
//
#include <hip/hip_runtime.h>
#include <math.h>

#define USE_TDM 1   // 1: tensor_load_to_lds double-buffered staging; 0: manual global->LDS

// ---------------------------------------------------------------------------
// Types for CDNA5 WMMA / TDM
// ---------------------------------------------------------------------------
typedef __attribute__((ext_vector_type(16))) __bf16   v16bf;
typedef __attribute__((ext_vector_type(8)))  float    v8f;
typedef __attribute__((ext_vector_type(4)))  unsigned v4u;
typedef __attribute__((ext_vector_type(8)))  unsigned v8u;

union Frag {
    v16bf v;
    float4 f[2];
};

__device__ __forceinline__ float wave_sum32(float x) {
#pragma unroll
    for (int off = 16; off > 0; off >>= 1) x += __shfl_xor(x, off, 32);
    return x;
}

__device__ __forceinline__ void st_bf16x4(__bf16* p, float a, float b, float c, float d) {
    p[0] = (__bf16)a; p[1] = (__bf16)b; p[2] = (__bf16)c; p[3] = (__bf16)d;
}

// ---------------------------------------------------------------------------
// TDM: issue a 2D tile load (rows x tile_k bf16 elements, row stride k_elems)
// from global memory into LDS at lds_off. Tile rows land contiguously in LDS
// (rows * tile_k * 2 bytes), matching the [rows][32] bf16 staging layout.
// Descriptor per CDNA5 ISA ch.8 (D# groups); groups 2/3 zeroed (2D tile).
// ---------------------------------------------------------------------------
__device__ __forceinline__ void tdm_load_2d(unsigned lds_off,
                                            unsigned long long gaddr,
                                            unsigned k_elems, unsigned rows,
                                            unsigned tile_k) {
    v4u g0;
    g0[0] = 1u;                                            // count=1, user mode
    g0[1] = lds_off;                                       // lds_addr (bytes)
    g0[2] = (unsigned)(gaddr & 0xffffffffu);               // global_addr[31:0]
    g0[3] = (unsigned)((gaddr >> 32) & 0x01ffffffu)        // global_addr[56:32]
            | (2u << 30);                                  // type=2 ("image")
    v8u g1;
    g1[0] = 1u << 16;                                      // data_size=1 -> 2 bytes
    g1[1] = (k_elems & 0xffffu) << 16;                     // tensor_dim0[15:0]
    g1[2] = ((k_elems >> 16) & 0xffffu)                    // tensor_dim0[31:16]
            | ((rows & 0xffffu) << 16);                    // tensor_dim1[15:0]
    g1[3] = ((rows >> 16) & 0xffffu)                       // tensor_dim1[31:16]
            | ((tile_k & 0xffffu) << 16);                  // tile_dim0
    g1[4] = rows & 0xffffu;                                // tile_dim1 (tile_dim2=0)
    g1[5] = k_elems;                                       // tensor_dim0_stride[31:0]
    g1[6] = 0u;                                            // stride hi, dim1_stride lo
    g1[7] = 0u;
    v4u gz = {0u, 0u, 0u, 0u};
    asm volatile("tensor_load_to_lds %0, %1, %2, %3"
                 :: "s"(g0), "s"(g1), "s"(gz), "s"(gz)
                 : "memory");
}

// ---------------------------------------------------------------------------
// Weight prep: W [Kin, N] f32 row-major  ->  Wt [N, Kpad] bf16 (zero padded K)
// ---------------------------------------------------------------------------
__global__ void pad_transpose_w(const float* __restrict__ W, __bf16* __restrict__ Wt,
                                int Kin, int Kpad, int N) {
    int idx = blockIdx.x * 256 + threadIdx.x;
    if (idx >= N * Kpad) return;
    int n = idx / Kpad;
    int k = idx - n * Kpad;
    float v = (k < Kin) ? W[(size_t)k * N + n] : 0.0f;
    Wt[idx] = (__bf16)v;
}

// ---------------------------------------------------------------------------
// Edge input build: A1[e] = [src(128), tgt(128), diff(128), d2, d, ef(128), pad(30)]
// One wave (32 lanes) per edge; 4 features per lane.
// ---------------------------------------------------------------------------
__global__ void build_edge_A(const float* __restrict__ nodef,
                             const long long* __restrict__ eidx,
                             const float* __restrict__ edgef,
                             __bf16* __restrict__ A1, int E) {
    int wave = threadIdx.x >> 5;
    int lane = threadIdx.x & 31;
    int e = blockIdx.x * 8 + wave;
    if (e >= E) return;

    long long r = eidx[e];        // source node
    long long c = eidx[E + e];    // target node
    const float4 s4 = *reinterpret_cast<const float4*>(nodef + (size_t)r * 128 + lane * 4);
    const float4 t4 = *reinterpret_cast<const float4*>(nodef + (size_t)c * 128 + lane * 4);
    float dx = s4.x - t4.x, dy = s4.y - t4.y, dz = s4.z - t4.z, dw = s4.w - t4.w;
    float part = dx * dx + dy * dy + dz * dz + dw * dw;
    float d2 = wave_sum32(part);
    float d = sqrtf(d2);

    __bf16* row = A1 + (size_t)e * 544;
    st_bf16x4(row + lane * 4,        s4.x, s4.y, s4.z, s4.w);
    st_bf16x4(row + 128 + lane * 4,  t4.x, t4.y, t4.z, t4.w);
    st_bf16x4(row + 256 + lane * 4,  dx, dy, dz, dw);
    const float4 e4 = *reinterpret_cast<const float4*>(edgef + (size_t)e * 128 + lane * 4);
    st_bf16x4(row + 386 + lane * 4,  e4.x, e4.y, e4.z, e4.w);
    if (lane == 0) { row[384] = (__bf16)d2; row[385] = (__bf16)d; }
    if (lane < 30) row[514 + lane] = (__bf16)0.0f;
}

// ---------------------------------------------------------------------------
// Fused GEMM (bf16 WMMA, f32 accumulate) + bias + LayerNorm(N) + exact GELU.
//   A  : [Mpad, K] bf16 row-major (Mpad multiple of 32, K = KS*32)
//   Bt : [N, K]  bf16 row-major (weights pre-transposed)
//   out: bf16 [*, N] (intermediate) or f32 [*, N] (final), rows < Mreal stored
// Block: 256 threads = 8 waves; tile BM=32, BN=N.
//   wave w -> m-tile (w&1), n-tile group (w>>1) covering J = NT/4 tiles.
// Staging: double-buffered LDS K-slices filled by the Tensor Data Mover
// (wave 0 issues tensor_load_to_lds; TENSORcnt + barrier publish the buffer),
// unioned with the C/LayerNorm staging area. All fragment loads of a K-slice
// are issued before the back-to-back WMMA chain so LDS reads overlap math.
// ---------------------------------------------------------------------------
template <int NT, int KS, bool OUT_BF16>
__global__ __launch_bounds__(256) void gemm_ln_gelu(
        const __bf16* __restrict__ A, const __bf16* __restrict__ Bt,
        const float* __restrict__ bias, const float* __restrict__ gamma,
        const float* __restrict__ beta, void* __restrict__ outp, int Mreal) {
    constexpr int N  = NT * 16;
    constexpr int K  = KS * 32;
    constexpr int J  = NT / 4;     // 16x16 output tiles per wave
    constexpr int NE = N / 32;     // LN elements per lane per row

    struct InT {
        alignas(16) __bf16 a[2][32 * 32];
        alignas(16) __bf16 b[2][N * 32];
    };
    union ShT {
        InT   in;
        float c[32 * N];
    };
    __shared__ ShT sh;

    const int t    = threadIdx.x;
    const int wave = t >> 5;
    const int lane = t & 31;
    const int mt   = wave & 1;      // m tile 0..1
    const int ng   = wave >> 1;     // n group 0..3
    const int m0   = blockIdx.x * 32;

    v8f acc[J];
#pragma unroll
    for (int j = 0; j < J; ++j) acc[j] = v8f{0.f,0.f,0.f,0.f,0.f,0.f,0.f,0.f};

    const int klo  = (lane < 16) ? 0 : 8;  // WMMA 16-bit A/B lane K striping
    const int arow = mt * 16 + (lane & 15);

#if USE_TDM
    const bool issuer = (wave == 0);       // wave-uniform: EXEC untouched
    const unsigned ldsA0 = (unsigned)(uintptr_t)(&sh.in.a[0][0]);
    const unsigned ldsA1 = (unsigned)(uintptr_t)(&sh.in.a[1][0]);
    const unsigned ldsB0 = (unsigned)(uintptr_t)(&sh.in.b[0][0]);
    const unsigned ldsB1 = (unsigned)(uintptr_t)(&sh.in.b[1][0]);
    const unsigned long long gA = (unsigned long long)(uintptr_t)A
                                + (unsigned long long)m0 * K * 2ull;
    const unsigned long long gB = (unsigned long long)(uintptr_t)Bt;

    if (issuer) {
        tdm_load_2d(ldsA0, gA, K, 32u, 32u);
        tdm_load_2d(ldsB0, gB, K, (unsigned)N, 32u);
    }
    for (int ks = 0; ks < KS; ++ks) {
        const int buf = ks & 1;
        if (issuer) __builtin_amdgcn_s_wait_tensorcnt(0);
        __syncthreads();                       // slice ks visible to all waves
        if (issuer && (ks + 1) < KS) {         // prefetch slice ks+1 via TDM
            const unsigned long long koff = (unsigned long long)(ks + 1) * 64ull;
            tdm_load_2d(buf ? ldsA0 : ldsA1, gA + koff, K, 32u, 32u);
            tdm_load_2d(buf ? ldsB0 : ldsB1, gB + koff, K, (unsigned)N, 32u);
        }
        // Batch all fragment loads, then back-to-back WMMAs.
        Frag a;
        a.f[0] = *reinterpret_cast<const float4*>(&sh.in.a[buf][arow * 32 + klo]);
        a.f[1] = *reinterpret_cast<const float4*>(&sh.in.a[buf][arow * 32 + klo + 16]);
        Frag b[J];
#pragma unroll
        for (int j = 0; j < J; ++j) {
            const int ncol = (ng * J + j) * 16 + (lane & 15);
            b[j].f[0] = *reinterpret_cast<const float4*>(&sh.in.b[buf][ncol * 32 + klo]);
            b[j].f[1] = *reinterpret_cast<const float4*>(&sh.in.b[buf][ncol * 32 + klo + 16]);
        }
#pragma unroll
        for (int j = 0; j < J; ++j) {
            acc[j] = __builtin_amdgcn_wmma_f32_16x16x32_bf16(
                false, a.v, false, b[j].v, (short)0, acc[j], false, false);
        }
        __syncthreads();                       // all reads of buf done
    }
#else
    for (int ks = 0; ks < KS; ++ks) {
        const int kbase = ks * 32;
        if (t < 128) {
            int row = t >> 2, grp = t & 3;
            *reinterpret_cast<float4*>(&sh.in.a[0][row * 32 + grp * 8]) =
                *reinterpret_cast<const float4*>(A + (size_t)(m0 + row) * K + kbase + grp * 8);
        }
        for (int idx = t; idx < N * 4; idx += 256) {
            int n = idx >> 2, grp = idx & 3;
            *reinterpret_cast<float4*>(&sh.in.b[0][n * 32 + grp * 8]) =
                *reinterpret_cast<const float4*>(Bt + (size_t)n * K + kbase + grp * 8);
        }
        __syncthreads();
        Frag a;
        a.f[0] = *reinterpret_cast<const float4*>(&sh.in.a[0][arow * 32 + klo]);
        a.f[1] = *reinterpret_cast<const float4*>(&sh.in.a[0][arow * 32 + klo + 16]);
        Frag b[J];
#pragma unroll
        for (int j = 0; j < J; ++j) {
            const int ncol = (ng * J + j) * 16 + (lane & 15);
            b[j].f[0] = *reinterpret_cast<const float4*>(&sh.in.b[0][ncol * 32 + klo]);
            b[j].f[1] = *reinterpret_cast<const float4*>(&sh.in.b[0][ncol * 32 + klo + 16]);
        }
#pragma unroll
        for (int j = 0; j < J; ++j) {
            acc[j] = __builtin_amdgcn_wmma_f32_16x16x32_bf16(
                false, a.v, false, b[j].v, (short)0, acc[j], false, false);
        }
        __syncthreads();
    }
#endif

    // Spill accumulators to LDS C staging (16x16 f32 C layout: lane<16 -> M=vg,
    // lane>=16 -> M=vg+8; N = lane&15 within tile).
    const int rbase = mt * 16 + ((lane >> 4) << 3);
#pragma unroll
    for (int j = 0; j < J; ++j) {
        const int col = (ng * J + j) * 16 + (lane & 15);
#pragma unroll
        for (int vg = 0; vg < 8; ++vg) {
            sh.c[(rbase + vg) * N + col] = acc[j][vg];
        }
    }
    __syncthreads();

    // Epilogue: bias + LayerNorm over N + exact GELU.  4 rows per wave.
    const float invN = 1.0f / (float)N;
#pragma unroll
    for (int rr = 0; rr < 4; ++rr) {
        const int r = wave * 4 + rr;
        float x[NE];
        float s = 0.f, sq = 0.f;
#pragma unroll
        for (int i = 0; i < NE; ++i) {
            const int c = lane + i * 32;
            float xv = sh.c[r * N + c] + bias[c];
            x[i] = xv; s += xv; sq += xv * xv;
        }
        s  = wave_sum32(s);
        sq = wave_sum32(sq);
        const float mean = s * invN;
        const float var  = sq * invN - mean * mean;
        const float rstd = rsqrtf(var + 1e-5f);
        const size_t grow = (size_t)m0 + (size_t)r;
        const bool wr = grow < (size_t)Mreal;
#pragma unroll
        for (int i = 0; i < NE; ++i) {
            const int c = lane + i * 32;
            float y = (x[i] - mean) * rstd * gamma[c] + beta[c];
            float g = 0.5f * y * (1.0f + erff(y * 0.70710678118654752f));
            if (wr) {
                if constexpr (OUT_BF16)
                    reinterpret_cast<__bf16*>(outp)[grow * N + c] = (__bf16)g;
                else
                    reinterpret_cast<float*>(outp)[grow * N + c] = g;
            }
        }
    }
}

// ---------------------------------------------------------------------------
// Aggregation helpers
// ---------------------------------------------------------------------------
__global__ void zero_f32(float* __restrict__ p, int n) {
    int idx = blockIdx.x * 256 + threadIdx.x;
    if (idx < n) p[idx] = 0.0f;
}

__global__ void scatter_add(const float* __restrict__ edge_out,
                            const long long* __restrict__ eidx,
                             float* __restrict__ agg, int E) {
    int idx = blockIdx.x * 256 + threadIdx.x;
    if (idx >= E * 32) return;
    int e = idx >> 5;
    int grp = idx & 31;
    int col = (int)eidx[E + e];
    const float4 v = *reinterpret_cast<const float4*>(edge_out + (size_t)e * 128 + grp * 4);
    float* dst = agg + (size_t)col * 128 + grp * 4;
    atomicAdd(dst + 0, v.x);
    atomicAdd(dst + 1, v.y);
    atomicAdd(dst + 2, v.z);
    atomicAdd(dst + 3, v.w);
}

__global__ void build_node_A(const float* __restrict__ nodef,
                             const float* __restrict__ agg,
                             __bf16* __restrict__ A2, int Nreal) {
    int idx = blockIdx.x * 256 + threadIdx.x;   // over Mpad*256
    int r = idx >> 8;
    int c = idx & 255;
    float v = 0.0f;
    if (r < Nreal) v = (c < 128) ? nodef[(size_t)r * 128 + c]
                                 : agg[(size_t)r * 128 + (c - 128)];
    A2[idx] = (__bf16)v;
}

// ---------------------------------------------------------------------------
// Host-side orchestration
// ---------------------------------------------------------------------------
extern "C" void kernel_launch(void* const* d_in, const int* in_sizes, int n_in,
                              void* d_out, int out_size, void* d_ws, size_t ws_size,
                              hipStream_t stream) {
    (void)in_sizes; (void)n_in; (void)out_size; (void)ws_size;

    constexpr int E = 320000, NN = 25000, Mpad2 = 25024;

    const float*      nodef  = (const float*)d_in[0];
    const long long*  eidx   = (const long long*)d_in[1];
    const float*      edgef  = (const float*)d_in[2];
    const float* eW1 = (const float*)d_in[3];
    const float* eb1 = (const float*)d_in[4];
    const float* eg1 = (const float*)d_in[5];
    const float* ebt1= (const float*)d_in[6];
    const float* eW2 = (const float*)d_in[7];
    const float* eb2 = (const float*)d_in[8];
    const float* eg2 = (const float*)d_in[9];
    const float* ebt2= (const float*)d_in[10];
    const float* nW1 = (const float*)d_in[11];
    const float* nb1 = (const float*)d_in[12];
    const float* ng1 = (const float*)d_in[13];
    const float* nbt1= (const float*)d_in[14];
    const float* nW2 = (const float*)d_in[15];
    const float* nb2 = (const float*)d_in[16];
    const float* ng2 = (const float*)d_in[17];
    const float* nbt2= (const float*)d_in[18];

    char* ws = (char*)d_ws;
    size_t off = 0;
    auto carve = [&](size_t bytes) -> char* {
        char* p = ws + off;
        off += (bytes + 255) & ~(size_t)255;
        return p;
    };
    __bf16* Wt1  = (__bf16*)carve((size_t)384 * 544 * 2);
    __bf16* Wt2  = (__bf16*)carve((size_t)128 * 384 * 2);
    __bf16* nWt1 = (__bf16*)carve((size_t)384 * 256 * 2);
    __bf16* nWt2 = (__bf16*)carve((size_t)128 * 384 * 2);
    __bf16* A1   = (__bf16*)carve((size_t)E * 544 * 2);
    __bf16* H1   = (__bf16*)carve((size_t)E * 384 * 2);
    float*  agg  = (float*)carve((size_t)NN * 128 * 4);
    __bf16* A2   = (__bf16*)carve((size_t)Mpad2 * 256 * 2);
    __bf16* H2   = (__bf16*)carve((size_t)Mpad2 * 384 * 2);

    float* node_out = (float*)d_out;
    float* edge_out = (float*)d_out + (size_t)NN * 128;

    // Weight conversion/transpose (tiny, stays in L2)
    pad_transpose_w<<<816, 256, 0, stream>>>(eW1, Wt1, 514, 544, 384);
    pad_transpose_w<<<192, 256, 0, stream>>>(eW2, Wt2, 384, 384, 128);
    pad_transpose_w<<<384, 256, 0, stream>>>(nW1, nWt1, 256, 256, 384);
    pad_transpose_w<<<192, 256, 0, stream>>>(nW2, nWt2, 384, 384, 128);

    // Edge pipeline
    build_edge_A<<<E / 8, 256, 0, stream>>>(nodef, eidx, edgef, A1, E);
    gemm_ln_gelu<24, 17, true ><<<E / 32, 256, 0, stream>>>(A1, Wt1, eb1, eg1, ebt1, H1, E);
    gemm_ln_gelu< 8, 12, false><<<E / 32, 256, 0, stream>>>(H1, Wt2, eb2, eg2, ebt2, edge_out, E);

    // Scatter-sum into node aggregate
    zero_f32<<<(NN * 128 + 255) / 256, 256, 0, stream>>>(agg, NN * 128);
    scatter_add<<<(E * 32) / 256, 256, 0, stream>>>(edge_out, eidx, agg, E);

    // Node pipeline
    build_node_A<<<Mpad2, 256, 0, stream>>>(nodef, agg, A2, NN);
    gemm_ln_gelu<24,  8, true ><<<Mpad2 / 32, 256, 0, stream>>>(A2, nWt1, nb1, ng1, nbt1, H2, Mpad2);
    gemm_ln_gelu< 8, 12, false><<<Mpad2 / 32, 256, 0, stream>>>(H2, nWt2, nb2, ng2, nbt2, node_out, NN);
}
